// TGCNet_16338055594467
// MI455X (gfx1250) — matrix operand order repaired and checked by reading, not compiled, not run on hardware
//
#include <hip/hip_runtime.h>
#include <hip/hip_bf16.h>
#include <math.h>

typedef __attribute__((ext_vector_type(8)))  float    v8f;
typedef __attribute__((ext_vector_type(16))) _Float16 v16h;
typedef __attribute__((ext_vector_type(8)))  _Float16 v8h;

// ---------------------------------------------------------------------------
// K1: deg[i] = 1.0 (self-loop weight; reference adds ones for every node)
// ---------------------------------------------------------------------------
__global__ void k_init_deg(float* __restrict__ deg, int N) {
    int i = blockIdx.x * blockDim.x + threadIdx.x;
    if (i < N) deg[i] = 1.0f;
}

// ---------------------------------------------------------------------------
// K2: deg[dst[e]] += ew[e]   (hardware fp32 atomics)
// ---------------------------------------------------------------------------
__global__ void k_accum_deg(float* __restrict__ deg,
                            const int* __restrict__ dst,
                            const float* __restrict__ ew, int E) {
    int e = blockIdx.x * blockDim.x + threadIdx.x;
    if (e < E) unsafeAtomicAdd(&deg[dst[e]], ew[e]);
}

// ---------------------------------------------------------------------------
// K3a: dinv[i] = rsqrt(deg[i])  (in place; deg >= 1 always due to self loops)
// ---------------------------------------------------------------------------
__global__ void k_dinv(float* __restrict__ deg, int N) {
    int i = blockIdx.x * blockDim.x + threadIdx.x;
    if (i < N) {
        float d = deg[i];
        deg[i] = (d > 0.0f) ? rsqrtf(d) : 0.0f;
    }
}

// ---------------------------------------------------------------------------
// K3b: self-loop init of aggregate: A[i][:] = x[i][:] * dinv[i]^2
//      one thread per float4 (fully coalesced, also initializes A)
// ---------------------------------------------------------------------------
__global__ void k_selfloop(float* __restrict__ A, const float* __restrict__ x,
                           const float* __restrict__ dinv, int N) {
    int t = blockIdx.x * blockDim.x + threadIdx.x;       // over N*16 float4s
    if (t >= N * 16) return;
    int n = t >> 4;
    float di = dinv[n];
    float w  = di * di;
    const float4 xx = ((const float4*)x)[t];
    float4 o;
    o.x = xx.x * w; o.y = xx.y * w; o.z = xx.z * w; o.w = xx.w * w;
    ((float4*)A)[t] = o;
}

// ---------------------------------------------------------------------------
// K4: edge aggregation  A[dst] += x[src] * (dinv[src]*ew*dinv[dst])
//     16 threads per edge, each handles one float4 of the 64-channel row.
//     ~410 MB of traffic total -> the bandwidth-dominant kernel (~18 us ideal)
// ---------------------------------------------------------------------------
__global__ void k_edge_agg(float* __restrict__ A, const float* __restrict__ x,
                           const float* __restrict__ dinv,
                           const int* __restrict__ src, const int* __restrict__ dst,
                           const float* __restrict__ ew, int E) {
    int t = blockIdx.x * blockDim.x + threadIdx.x;       // over E*16
    if (t >= E * 16) return;
    int e = t >> 4;
    int q = t & 15;
    int s = src[e];
    int d = dst[e];
    float w = dinv[s] * ew[e] * dinv[d];
    const float4 xs = *(const float4*)(x + (size_t)s * 64 + q * 4);
    float* Ad = A + (size_t)d * 64 + q * 4;
    unsafeAtomicAdd(Ad + 0, xs.x * w);
    unsafeAtomicAdd(Ad + 1, xs.y * w);
    unsafeAtomicAdd(Ad + 2, xs.z * w);
    unsafeAtomicAdd(Ad + 3, xs.w * w);
}

// ---------------------------------------------------------------------------
// K5: fold chained linears (tiny): Mz = conv_z_W @ lin_z_W[0:64],
//     bz' = conv_z_b @ lin_z_W[0:64] + lin_z_b ; same for the h gate.
// ---------------------------------------------------------------------------
__global__ void k_combine_weights(const float* __restrict__ czW, const float* __restrict__ czb,
                                  const float* __restrict__ lzW, const float* __restrict__ lzb,
                                  const float* __restrict__ chW, const float* __restrict__ chb,
                                  const float* __restrict__ lhW, const float* __restrict__ lhb,
                                  float* __restrict__ Wz, float* __restrict__ bz,
                                  float* __restrict__ Wh, float* __restrict__ bh) {
    int t = threadIdx.x;                                  // 256 threads
    for (int idx = t; idx < 64 * 64; idx += 256) {
        int i = idx >> 6, j = idx & 63;
        float sz = 0.0f, sh = 0.0f;
        for (int k = 0; k < 64; ++k) {
            sz += czW[i * 64 + k] * lzW[k * 64 + j];      // lin rows 0..63 = gcn half
            sh += chW[i * 64 + k] * lhW[k * 64 + j];
        }
        Wz[idx] = sz;
        Wh[idx] = sh;
    }
    if (t < 64) {
        float sz = 0.0f, sh = 0.0f;
        for (int k = 0; k < 64; ++k) {
            sz += czb[k] * lzW[k * 64 + t];
            sh += chb[k] * lhW[k * 64 + t];
        }
        bz[t] = sz + lzb[t];
        bh[t] = sh + lhb[t];
    }
}

// ---------------------------------------------------------------------------
// K6: per-16-node-tile WMMA kernel.
//   P = A_tile @ Wz + bz ; Q = A_tile @ Wh + bh
//   H_new = (1 - sigmoid(P)) * tanh(Q)      (H==0 simplification)
//         = tanh(Q) / (1 + exp(P))          branch-free, native exp/rcp
//   out_tile = relu(H_new) @ head_W(padded 64x16) + head_b
// One wave per tile; 4 waves per block; weights live transposed (f16) in LDS.
// ---------------------------------------------------------------------------
#define TPB_TILES 4
__global__ __launch_bounds__(128) void k_node_wmma(
    const float* __restrict__ Aagg,
    const float* __restrict__ Wz, const float* __restrict__ bz,
    const float* __restrict__ Wh, const float* __restrict__ bh,
    const float* __restrict__ headW, const float* __restrict__ headb,
    float* __restrict__ out, int N, int numTiles)
{
    __shared__ __attribute__((aligned(32))) _Float16 sWzT[64 * 64];  // [col][k]
    __shared__ __attribute__((aligned(32))) _Float16 sWhT[64 * 64];  // [col][k]
    __shared__ __attribute__((aligned(32))) _Float16 sWoT[16 * 64];  // [col][k], cols 8..15 zero
    __shared__ float sBz[64], sBh[64], sBo[16];
    __shared__ __attribute__((aligned(32))) _Float16 sS[TPB_TILES][16 * 64]; // relu(H) staging

    const int t = threadIdx.x;
    for (int idx = t; idx < 64 * 64; idx += 128) {
        int j = idx >> 6, k = idx & 63;                   // transposed: [j][k] = W[k][j]
        sWzT[idx] = (_Float16)Wz[k * 64 + j];
        sWhT[idx] = (_Float16)Wh[k * 64 + j];
    }
    for (int idx = t; idx < 16 * 64; idx += 128) {
        int j = idx >> 6, k = idx & 63;
        sWoT[idx] = (j < 8) ? (_Float16)headW[k * 8 + j] : (_Float16)0.0f;
    }
    if (t < 64) { sBz[t] = bz[t]; sBh[t] = bh[t]; }
    if (t < 16) { sBo[t] = (t < 8) ? headb[t] : 0.0f; }
    __syncthreads();

    const int wave  = t >> 5;
    const int lane  = t & 31;
    const int half_ = lane >> 4;          // which 16-lane half
    const int r     = lane & 15;          // A row within tile == B/C column index
    const int tile  = blockIdx.x * TPB_TILES + wave;
    if (tile >= numTiles) return;
    const int base = tile * 16;
    const int rowg = min(base + r, N - 1);

    // --- A fragments (16x32 f16 layout): i -> K=ks*32+16*half+i
    v16h afrag[2];
    const float* Arow = Aagg + (size_t)rowg * 64;
#pragma unroll
    for (int ks = 0; ks < 2; ++ks) {
        const float* p0 = Arow + ks * 32 + 16 * half_;       // K run 0 (8 floats)
        const float* p1 = Arow + ks * 32 + 16 * half_ + 8;   // K run 1 (8 floats)
        float4 u0 = *(const float4*)(p0);
        float4 u1 = *(const float4*)(p0 + 4);
        float4 w0 = *(const float4*)(p1);
        float4 w1 = *(const float4*)(p1 + 4);
        v16h a;
        a[0]  = (_Float16)u0.x; a[1]  = (_Float16)u0.y; a[2]  = (_Float16)u0.z; a[3]  = (_Float16)u0.w;
        a[4]  = (_Float16)u1.x; a[5]  = (_Float16)u1.y; a[6]  = (_Float16)u1.z; a[7]  = (_Float16)u1.w;
        a[8]  = (_Float16)w0.x; a[9]  = (_Float16)w0.y; a[10] = (_Float16)w0.z; a[11] = (_Float16)w0.w;
        a[12] = (_Float16)w1.x; a[13] = (_Float16)w1.y; a[14] = (_Float16)w1.z; a[15] = (_Float16)w1.w;
        afrag[ks] = a;
    }

    _Float16* Sw = &sS[wave][0];

#pragma unroll
    for (int nt = 0; nt < 4; ++nt) {                       // output column tiles
        const int col = nt * 16 + r;
        const float bzv = sBz[col];
        const float bhv = sBh[col];
        v8f accP = {bzv, bzv, bzv, bzv, bzv, bzv, bzv, bzv};
        v8f accQ = {bhv, bhv, bhv, bhv, bhv, bhv, bhv, bhv};
#pragma unroll
        for (int ks = 0; ks < 2; ++ks) {                   // B 32x16 f16: lane i -> K=ks*32+16*half+i
            v16h b = *(const v16h*)&sWzT[col * 64 + ks * 32 + 16 * half_];
            accP = __builtin_amdgcn_wmma_f32_16x16x32_f16(
                false, afrag[ks], false, b, (short)0, accP, false, false);
        }
#pragma unroll
        for (int ks = 0; ks < 2; ++ks) {
            v16h b = *(const v16h*)&sWhT[col * 64 + ks * 32 + 16 * half_];
            accQ = __builtin_amdgcn_wmma_f32_16x16x32_f16(
                false, afrag[ks], false, b, (short)0, accQ, false, false);
        }
        // C/D layout: VGPR v holds (M = v + 8*half, Ncol = col)
        // H_new = tanh(q)/(1+e^p); tanh(q) = 1 - 2/(1+e^{2q}).  Branch-free:
        // native v_exp_f32 / v_rcp_f32, correct limits via IEEE inf semantics.
#pragma unroll
        for (int v = 0; v < 8; ++v) {
            float p  = accP[v];
            float q  = accQ[v];
            float ep  = __expf(p);
            float e2q = __expf(2.0f * q);
            float ht  = 1.0f - 2.0f * __builtin_amdgcn_rcpf(1.0f + e2q);
            float hn  = ht * __builtin_amdgcn_rcpf(1.0f + ep);
            float rl  = fmaxf(hn, 0.0f);
            Sw[(v + 8 * half_) * 64 + col] = (_Float16)rl;
        }
    }

    // --- head: [16,64] @ [64,16(pad)] ; A fragments from Sw (row-major f16)
    const float bov = sBo[r];
    v8f accO = {bov, bov, bov, bov, bov, bov, bov, bov};
#pragma unroll
    for (int ks = 0; ks < 2; ++ks) {
        v8h run0 = *(const v8h*)&Sw[r * 64 + ks * 32 + 16 * half_];
        v8h run1 = *(const v8h*)&Sw[r * 64 + ks * 32 + 16 * half_ + 8];
        v16h a;
#pragma unroll
        for (int i = 0; i < 8; ++i) { a[i] = run0[i]; a[8 + i] = run1[i]; }
        v16h b = *(const v16h*)&sWoT[r * 64 + ks * 32 + 16 * half_];
        accO = __builtin_amdgcn_wmma_f32_16x16x32_f16(
            false, a, false, b, (short)0, accO, false, false);
    }
    if (r < 8) {                                           // only 8 real output columns
#pragma unroll
        for (int v = 0; v < 8; ++v) {
            int m = base + v + 8 * half_;
            if (m < N) out[(size_t)m * 8 + r] = accO[v];
        }
    }
}

// ---------------------------------------------------------------------------
// Host launcher
// ---------------------------------------------------------------------------
extern "C" void kernel_launch(void* const* d_in, const int* in_sizes, int n_in,
                              void* d_out, int out_size, void* d_ws, size_t ws_size,
                              hipStream_t stream) {
    const int N = in_sizes[0] / 64;   // x: [N, 64]
    const int E = in_sizes[2];        // edge_weight: [E]

    const float* x   = (const float*)d_in[0];
    const int*   ei  = (const int*)d_in[1];      // [2,E]: src = ei, dst = ei+E
    const float* ew  = (const float*)d_in[2];
    const float* czW = (const float*)d_in[3];
    const float* czb = (const float*)d_in[4];
    // d_in[5], d_in[6] (conv_r), d_in[11], d_in[12] (lin_r): dead (H==0 -> H*R==0)
    const float* chW = (const float*)d_in[7];
    const float* chb = (const float*)d_in[8];
    const float* lzW = (const float*)d_in[9];
    const float* lzb = (const float*)d_in[10];
    const float* lhW = (const float*)d_in[13];
    const float* lhb = (const float*)d_in[14];
    const float* hW  = (const float*)d_in[15];
    const float* hb  = (const float*)d_in[16];

    const int* srcI = ei;
    const int* dstI = ei + E;

    // workspace layout (floats)
    float* ws   = (float*)d_ws;
    float* deg  = ws;                          // N       (becomes dinv in place)
    float* Aagg = ws + N;                      // N*64
    float* Wz   = Aagg + (size_t)N * 64;       // 4096
    float* Wh   = Wz + 4096;                   // 4096
    float* bz   = Wh + 4096;                   // 64
    float* bh   = bz + 64;                     // 64

    const int numTiles = (N + 15) / 16;

    k_init_deg<<<(N + 255) / 256, 256, 0, stream>>>(deg, N);
    k_accum_deg<<<(E + 255) / 256, 256, 0, stream>>>(deg, dstI, ew, E);
    k_dinv<<<(N + 255) / 256, 256, 0, stream>>>(deg, N);
    k_selfloop<<<(N * 16 + 255) / 256, 256, 0, stream>>>(Aagg, x, deg, N);
    k_edge_agg<<<(E * 16 + 255) / 256, 256, 0, stream>>>(Aagg, x, deg, srcI, dstI, ew, E);
    k_combine_weights<<<1, 256, 0, stream>>>(czW, czb, lzW, lzb, chW, chb, lhW, lhb,
                                             Wz, bz, Wh, bh);
    k_node_wmma<<<(numTiles + TPB_TILES - 1) / TPB_TILES, 128, 0, stream>>>(
        Aagg, Wz, bz, Wh, bh, hW, hb, (float*)d_out, N, numTiles);
}
